// Block_56856777064506
// MI455X (gfx1250) — matrix operand order repaired and checked by reading, not compiled, not run on hardware
//
#include <hip/hip_runtime.h>
#include <math.h>

// ---------------------------------------------------------------------------
// CDNA5 (gfx1250) transformer block: LN -> QKV gemm -> flash attn -> proj
// -> LN -> MLP (GELU) -> proj, all GEMMs on v_wmma_f32_16x16x32_f16.
// GEMM tiles are staged into LDS by the Tensor Data Mover (tensor_load_to_lds,
// TENSORcnt) when available, else by async global->LDS copies (ASYNCcnt).
// ---------------------------------------------------------------------------

typedef _Float16     v16h __attribute__((ext_vector_type(16)));
typedef _Float16     v8h  __attribute__((ext_vector_type(8)));
typedef float        v8f  __attribute__((ext_vector_type(8)));
typedef int          v4i  __attribute__((ext_vector_type(4)));
typedef int          v8i  __attribute__((ext_vector_type(8)));
typedef unsigned int v4u  __attribute__((ext_vector_type(4)));

#define AS1 __attribute__((address_space(1)))
#define AS3 __attribute__((address_space(3)))

#if __has_builtin(__builtin_amdgcn_global_load_async_to_lds_b128)
#define HAVE_ASYNC_LDS 1
#else
#define HAVE_ASYNC_LDS 0
#endif

#if __has_builtin(__builtin_amdgcn_tensor_load_to_lds)
#define HAVE_TDM 1
#else
#define HAVE_TDM 0
#endif

// Copy 16 bytes global -> LDS (async when supported).
__device__ __forceinline__ void copy16_g2l(const _Float16* g, _Float16* l) {
#if HAVE_ASYNC_LDS
  __builtin_amdgcn_global_load_async_to_lds_b128((AS1 v4i*)g, (AS3 v4i*)l, 0, 0);
#else
  *(v8h*)l = *(const v8h*)g;
#endif
}
__device__ __forceinline__ void g2l_wait() {
#if HAVE_ASYNC_LDS
  asm volatile("s_wait_asynccnt 0" ::: "memory");
#endif
}

__device__ __forceinline__ void tensor_wait() {
#if __has_builtin(__builtin_amdgcn_s_wait_tensorcnt)
  __builtin_amdgcn_s_wait_tensorcnt((short)0);
#else
  asm volatile("s_wait_tensorcnt 0" ::: "memory");
#endif
}

#if HAVE_TDM
// Issue one TDM 2D tile load: 128 rows x 32 f16 from global (row stride
// `stride_elems`) into LDS at `lds_ptr`, with hardware padding of 4 DWORDs
// after every 16 DWORDs stored -> LDS row stride of 40 halves.
__device__ __forceinline__ void tdm_load_tile(const _Float16* g,
                                              const _Float16* lds_ptr,
                                              int stride_elems) {
  unsigned long long ga = (unsigned long long)g;
  unsigned lds_addr = (unsigned)(unsigned long long)(AS3 _Float16*)lds_ptr;
  v4u g0;
  g0.x = 1u;                                   // count=1, user descriptor
  g0.y = lds_addr;                             // D#.lds_addr
  g0.z = (unsigned)ga;                         // global_addr[31:0]
  g0.w = (unsigned)((ga >> 32) & 0x1ffffffull) // global_addr[56:32]
       | (2u << 30);                           // type = 2 ("image")
  v8i g1;
  g1[0] = (int)((1u << 16)    // data_size = 1 -> 2 bytes
              | (1u << 20)    // pad_enable
              | (3u << 22)    // pad_interval: 16 DWORDs
              | (3u << 25));  // pad_amount:   4 DWORDs
  g1[1] = (int)(32u << 16);   // tensor_dim0 = 32 (bits 79:48, low half)
  g1[2] = (int)(128u << 16);  // tensor_dim0 hi = 0 ; tensor_dim1 = 128
  g1[3] = (int)(32u << 16);   // tensor_dim1 hi = 0 ; tile_dim0 = 32
  g1[4] = 128;                // tile_dim1 = 128 ; tile_dim2 = 0
  g1[5] = stride_elems;       // tensor_dim0_stride[31:0]
  g1[6] = 0;                  // stride hi, dim1_stride lo
  g1[7] = 0;
  v4i z4 = {};
#if __clang_major__ >= 23
  v8i z8 = {};
  __builtin_amdgcn_tensor_load_to_lds(g0, g1, z4, z4, z8, 0);
#else
  __builtin_amdgcn_tensor_load_to_lds(g0, g1, z4, z4, 0);
#endif
}
#endif

__device__ __forceinline__ v8f wmma_f16(v16h a, v16h b, v8f c) {
  // D = A(16x32 f16) * B(32x16 f16) + C(16x16 f32)
  return __builtin_amdgcn_wmma_f32_16x16x32_f16(
      /*neg_a=*/false, a, /*neg_b=*/false, b,
      /*c_mod=*/(short)0, c, /*reuse_a=*/false, /*reuse_b=*/false);
}

__device__ __forceinline__ v16h comb16(v8h lo, v8h hi) {
  v16h r;
#pragma unroll
  for (int i = 0; i < 8; i++) { r[i] = lo[i]; r[i + 8] = hi[i]; }
  return r;
}

// A-fragment (16x32 f16) from row-major source (leading dim ld, elements).
// Contiguous k order (matches load_b_frag): lane l -> row m=l&15, h=l>>4; element e <-> K=16h+e.
// Two contiguous 16B loads per lane.
__device__ __forceinline__ v16h load_a_frag(const _Float16* base, int ld, int lane) {
  int m = lane & 15, h = lane >> 4;
  const _Float16* p = base + (size_t)m * ld + 16 * h;
  return comb16(*(const v8h*)p, *(const v8h*)(p + 8));
}

// B-fragment (32x16 f16) from B^T stored row-major [N,K] (leading dim ld=K).
// Contiguous k order (matches load_a_frag): lane l -> col n=l&15, h=l>>4; element e <-> K=16h+e.
__device__ __forceinline__ v16h load_b_frag(const _Float16* baseT, int ld, int lane) {
  int n = lane & 15, h = lane >> 4;
  return *(const v16h*)(baseT + (size_t)n * ld + 16 * h);
}

// ---------------------------------------------------------------------------
// LayerNorm: one block per row of D=1024; outputs fp32 + f16 copies.
// ---------------------------------------------------------------------------
__global__ __launch_bounds__(256) void ln_kernel(
    const float* __restrict__ x, const float* __restrict__ w,
    const float* __restrict__ b, float* __restrict__ y32,
    _Float16* __restrict__ y16, int D) {
  int row = blockIdx.x;
  const float* xr = x + (size_t)row * D;
  float s = 0.f, ss = 0.f;
  for (int i = threadIdx.x; i < D; i += blockDim.x) {
    float v = xr[i];
    s += v; ss += v * v;
  }
#pragma unroll
  for (int off = 16; off > 0; off >>= 1) {
    s  += __shfl_down(s, off, 32);
    ss += __shfl_down(ss, off, 32);
  }
  __shared__ float sh[2][8];
  int wid = threadIdx.x >> 5, lane = threadIdx.x & 31;
  if (lane == 0) { sh[0][wid] = s; sh[1][wid] = ss; }
  __syncthreads();
  if (threadIdx.x == 0) {
    float a = 0.f, c = 0.f;
#pragma unroll
    for (int i = 0; i < 8; i++) { a += sh[0][i]; c += sh[1][i]; }
    sh[0][0] = a; sh[1][0] = c;
  }
  __syncthreads();
  float mean = sh[0][0] / (float)D;
  float var  = sh[1][0] / (float)D - mean * mean;
  float inv  = rsqrtf(var + 1e-5f);
  for (int i = threadIdx.x; i < D; i += blockDim.x) {
    float v = (xr[i] - mean) * inv * w[i] + b[i];
    y32[(size_t)row * D + i] = v;
    y16[(size_t)row * D + i] = (_Float16)v;
  }
}

// ---------------------------------------------------------------------------
// Weight convert + transpose: Wt[n*K + k] = (f16) W[k*N + n]
// ---------------------------------------------------------------------------
__global__ __launch_bounds__(256) void wcvt_kernel(
    const float* __restrict__ W, _Float16* __restrict__ Wt, int K, int N) {
  size_t idx = (size_t)blockIdx.x * blockDim.x + threadIdx.x;
  size_t total = (size_t)K * N;
  if (idx >= total) return;
  size_t k = idx % (size_t)K;
  size_t n = idx / (size_t)K;
  Wt[idx] = (_Float16)W[k * (size_t)N + n];
}

// ---------------------------------------------------------------------------
// WMMA GEMM: C[M,N] = A[M,K](f16) * Wt[N,K]^T(f16) + bias, with epilogue.
// Block tile 128x128, 8 waves (4 along M x 2 along N), wave tile 32x64.
// Double-buffered LDS staging via TDM (wave 0 issues tensor_load_to_lds,
// hardware pads rows to 40 halves) or async global->LDS fallback.
// EPI 0: store f16 ; EPI 1: store f32 + residual ; EPI 2: GELU -> f16
// ---------------------------------------------------------------------------
#define LDS_STRIDE 40

template <int EPI>
__global__ __launch_bounds__(256) void gemm_wmma(
    const _Float16* __restrict__ A, const _Float16* __restrict__ Wt,
    const float* __restrict__ bias, const float* __restrict__ res,
    float* __restrict__ outf, _Float16* __restrict__ outh,
    int M, int N, int K) {
  __shared__ __align__(16) _Float16 As[2][128 * LDS_STRIDE];
  __shared__ __align__(16) _Float16 Bs[2][128 * LDS_STRIDE];

  int tid  = threadIdx.x;
  int lane = tid & 31;
  int wid  = tid >> 5;
  int wm = wid & 3, wn = wid >> 2;        // 4 x 2 wave grid
  int m0 = blockIdx.y * 128;
  int n0 = blockIdx.x * 128;

#if HAVE_TDM
  // Wave 0 drives the Tensor Data Mover for both tiles; TDM ignores EXEC and
  // issues once per wave. Other waves sync on the workgroup barrier.
  if (wid == 0) {
    tdm_load_tile(A  + (size_t)m0 * K, &As[0][0], K);
    tdm_load_tile(Wt + (size_t)n0 * K, &Bs[0][0], K);
    tensor_wait();
  }
  __syncthreads();
#else
  // staging assignment: thread -> (row, 16-col half) of the 128x32 tile
  int sr = tid >> 1;
  int sc = (tid & 1) * 16;
  const _Float16* gA = A  + (size_t)(m0 + sr) * K + sc;
  const _Float16* gB = Wt + (size_t)(n0 + sr) * K + sc;
  _Float16* lA0 = &As[0][sr * LDS_STRIDE + sc];
  _Float16* lB0 = &Bs[0][sr * LDS_STRIDE + sc];
  _Float16* lA1 = &As[1][sr * LDS_STRIDE + sc];
  _Float16* lB1 = &Bs[1][sr * LDS_STRIDE + sc];
  copy16_g2l(gA, lA0);      copy16_g2l(gA + 8, lA0 + 8);
  copy16_g2l(gB, lB0);      copy16_g2l(gB + 8, lB0 + 8);
  g2l_wait();
  __syncthreads();
#endif

  v8f acc[2][4] = {};
  int nl = lane & 15, h = lane >> 4;

  for (int k0 = 0; k0 < K; k0 += 32) {
    int cur = (k0 >> 5) & 1;
    // issue staging of the next k-slice into the other buffer
    if (k0 + 32 < K) {
#if HAVE_TDM
      if (wid == 0) {
        tdm_load_tile(A  + (size_t)m0 * K + k0 + 32, &As[cur ^ 1][0], K);
        tdm_load_tile(Wt + (size_t)n0 * K + k0 + 32, &Bs[cur ^ 1][0], K);
      }
#else
      const _Float16* gA2 = gA + (k0 + 32);
      const _Float16* gB2 = gB + (k0 + 32);
      _Float16* lA = cur ? lA0 : lA1;
      _Float16* lB = cur ? lB0 : lB1;
      copy16_g2l(gA2, lA);  copy16_g2l(gA2 + 8, lA + 8);
      copy16_g2l(gB2, lB);  copy16_g2l(gB2 + 8, lB + 8);
#endif
    }
    // compute from LDS buffer `cur`
    v16h bfr[4];
#pragma unroll
    for (int tj = 0; tj < 4; tj++) {
      const _Float16* p = &Bs[cur][(wn * 64 + tj * 16 + nl) * LDS_STRIDE + 16 * h];
      bfr[tj] = comb16(*(const v8h*)p, *(const v8h*)(p + 8));
    }
#pragma unroll
    for (int ti = 0; ti < 2; ti++) {
      const _Float16* p = &As[cur][(wm * 32 + ti * 16 + nl) * LDS_STRIDE + 16 * h];
      v16h afr = comb16(*(const v8h*)p, *(const v8h*)(p + 8));
#pragma unroll
      for (int tj = 0; tj < 4; tj++)
        acc[ti][tj] = wmma_f16(afr, bfr[tj], acc[ti][tj]);
    }
#if HAVE_TDM
    if (wid == 0) tensor_wait();   // next slice landed in LDS
#else
    g2l_wait();
#endif
    __syncthreads();   // everyone done reading `cur` before it is overwritten
  }

#pragma unroll
  for (int ti = 0; ti < 2; ti++)
#pragma unroll
    for (int tj = 0; tj < 4; tj++) {
      int n = n0 + wn * 64 + tj * 16 + nl;
      float bn = bias[n];
#pragma unroll
      for (int r = 0; r < 8; r++) {
        int m = m0 + wm * 32 + ti * 16 + r + 8 * h;
        float v = acc[ti][tj][r] + bn;
        if (EPI == 1) {
          outf[(size_t)m * N + n] = v + res[(size_t)m * N + n];
        } else if (EPI == 2) {
          float g = 0.5f * v * (1.f + erff(v * 0.70710678118f));
          outh[(size_t)m * N + n] = (_Float16)g;
        } else {
          outh[(size_t)m * N + n] = (_Float16)v;
        }
      }
    }
}

// ---------------------------------------------------------------------------
// Flash attention (causal), one wave per 16-query tile, hd = 64.
// qkv layout: [B*S, 3*D] f16; output o: [B*S, D] f16.
// ---------------------------------------------------------------------------
__global__ __launch_bounds__(32) void attn_kernel(
    const _Float16* __restrict__ qkv, _Float16* __restrict__ o,
    int S, int D, int H) {
  const int hd = 64;
  int lane = threadIdx.x;
  int qt = blockIdx.x;
  int bh = blockIdx.y;
  int b = bh / H, hh = bh % H;
  int m0 = qt * 16;
  int ld = 3 * D;
  const _Float16* qkvb = qkv + (size_t)b * S * ld;
  const _Float16* Q  = qkvb + hh * hd;
  const _Float16* Kp = qkvb + D + hh * hd;
  const _Float16* Vp = qkvb + 2 * D + hh * hd;

  __shared__ __align__(32) _Float16 Pl[16][32];
  __shared__ __align__(32) _Float16 Vl[32][64];

  int h = lane >> 4, nl = lane & 15;

  // Q fragments for K-dim steps 0 and 32 (hd = 64) -- fixed over key loop
  v16h aq0 = load_a_frag(Q + (size_t)m0 * ld + 0, ld, lane);
  v16h aq1 = load_a_frag(Q + (size_t)m0 * ld + 32, ld, lane);

  v8f o_acc[4] = {};
  float rmax[8], rsum[8];
#pragma unroll
  for (int r = 0; r < 8; r++) { rmax[r] = -1e30f; rsum[r] = 0.f; }

  const float scale = 0.125f;  // 1/sqrt(64)

  for (int s0 = 0; s0 < m0 + 16; s0 += 32) {
    // ---- scores S = Q K^T for two 16-key tiles ----
    v8f sc[2];
#pragma unroll
    for (int t = 0; t < 2; t++) {
      const _Float16* kb = Kp + (size_t)(s0 + t * 16) * ld;
      v16h b0 = load_b_frag(kb, ld, lane);       // hd 0..31
      v16h b1 = load_b_frag(kb + 32, ld, lane);  // hd 32..63
      v8f z = {};
      z = wmma_f16(aq0, b0, z);
      z = wmma_f16(aq1, b1, z);
      sc[t] = z;
    }
    // ---- scale + causal mask + per-chunk row max ----
    float cmax[8];
#pragma unroll
    for (int r = 0; r < 8; r++) {
      int m = m0 + r + 8 * h;
      float v0 = sc[0][r] * scale;
      float v1 = sc[1][r] * scale;
      if (s0 + nl > m)      v0 = -1e30f;
      if (s0 + 16 + nl > m) v1 = -1e30f;
      sc[0][r] = v0; sc[1][r] = v1;
      cmax[r] = fmaxf(v0, v1);
    }
#pragma unroll
    for (int off = 1; off < 16; off <<= 1)
#pragma unroll
      for (int r = 0; r < 8; r++)
        cmax[r] = fmaxf(cmax[r], __shfl_xor(cmax[r], off, 32));
    // ---- online softmax update ----
    float alpha[8];
#pragma unroll
    for (int r = 0; r < 8; r++) {
      float nm = fmaxf(rmax[r], cmax[r]);
      alpha[r] = __expf(rmax[r] - nm);
      rmax[r] = nm;
    }
    float csum[8];
#pragma unroll
    for (int r = 0; r < 8; r++) {
      float p0 = __expf(sc[0][r] - rmax[r]);
      float p1 = __expf(sc[1][r] - rmax[r]);
      sc[0][r] = p0; sc[1][r] = p1;
      csum[r] = p0 + p1;
    }
#pragma unroll
    for (int off = 1; off < 16; off <<= 1)
#pragma unroll
      for (int r = 0; r < 8; r++)
        csum[r] += __shfl_xor(csum[r], off, 32);
#pragma unroll
    for (int r = 0; r < 8; r++) rsum[r] = rsum[r] * alpha[r] + csum[r];
#pragma unroll
    for (int t = 0; t < 4; t++)
#pragma unroll
      for (int r = 0; r < 8; r++) o_acc[t][r] *= alpha[r];

    // ---- P tile (C-frag layout) -> LDS, row-major [16 q][32 k] ----
#pragma unroll
    for (int r = 0; r < 8; r++) {
      Pl[r + 8 * h][nl]      = (_Float16)sc[0][r];
      Pl[r + 8 * h][16 + nl] = (_Float16)sc[1][r];
    }
    // ---- stage V chunk [32 keys][64 hd] into LDS ----
    {
      const _Float16* g = Vp + (size_t)(s0 + lane) * ld;
#pragma unroll
      for (int j = 0; j < 4; j++)
        *(v16h*)&Vl[lane][j * 16] = *(const v16h*)(g + j * 16);
    }
    __syncthreads();  // single-wave WG: cheap; orders LDS for the compiler

    // ---- P as A-fragment (16x32) from LDS ----
    v16h ap;
    {
      const _Float16* p = &Pl[nl][16 * h];
      ap = comb16(*(const v8h*)p, *(const v8h*)(p + 8));
    }
    // ---- O += P V : four 16-col tiles of hd ----
#pragma unroll
    for (int t = 0; t < 4; t++) {
      v16h bv;
#pragma unroll
      for (int e = 0; e < 16; e++) bv[e] = Vl[16 * h + e][t * 16 + nl];
      o_acc[t] = wmma_f16(ap, bv, o_acc[t]);
    }
    __syncthreads();
  }

  // ---- finalize: O / rowsum -> f16 ----
#pragma unroll
  for (int t = 0; t < 4; t++)
#pragma unroll
    for (int r = 0; r < 8; r++) {
      int m = m0 + r + 8 * h;
      float v = o_acc[t][r] / rsum[r];
      o[((size_t)(b * S + m)) * D + hh * hd + t * 16 + nl] = (_Float16)v;
    }
}

// ---------------------------------------------------------------------------
// Host-side orchestration.
// ---------------------------------------------------------------------------
extern "C" void kernel_launch(void* const* d_in, const int* in_sizes, int n_in,
                              void* d_out, int out_size, void* d_ws, size_t ws_size,
                              hipStream_t stream) {
  (void)in_sizes; (void)n_in; (void)out_size; (void)ws_size;
  const int B = 2, S = 2048, D = 1024, H = 16;
  const int R = B * S;  // 4096 rows

  const float* x          = (const float*)d_in[0];
  const float* ln1_w      = (const float*)d_in[1];
  const float* ln1_b      = (const float*)d_in[2];
  const float* qkv_w      = (const float*)d_in[3];
  const float* qkv_b      = (const float*)d_in[4];
  const float* attn_out_w = (const float*)d_in[5];
  const float* attn_out_b = (const float*)d_in[6];
  const float* ln2_w      = (const float*)d_in[7];
  const float* ln2_b      = (const float*)d_in[8];
  const float* c_fc_w     = (const float*)d_in[9];
  const float* c_fc_b     = (const float*)d_in[10];
  const float* c_proj_w   = (const float*)d_in[11];
  const float* c_proj_b   = (const float*)d_in[12];

  // Workspace layout (bytes), lifetime-aliased; total 80 MB:
  //  [ 0,32M)  qkv16 (24M) + o16 (8M @24M)  -> reused as a16 (32M)
  //  [32,48M)  xn32 (fp32)                  -> reused as hn32
  //  [48,56M)  xn16 (f16)                   -> reused as hn16
  //  [56,72M)  h32  (fp32)
  //  [72,80M)  wT   (f16, reused per GEMM)
  char* ws = (char*)d_ws;
  const size_t MB = 1024ull * 1024ull;
  _Float16* qkv16 = (_Float16*)(ws + 0);
  _Float16* o16   = (_Float16*)(ws + 24 * MB);
  _Float16* a16   = (_Float16*)(ws + 0);
  float*    xn32  = (float*)(ws + 32 * MB);
  _Float16* xn16  = (_Float16*)(ws + 48 * MB);
  float*    h32   = (float*)(ws + 56 * MB);
  _Float16* wT    = (_Float16*)(ws + 72 * MB);

  // 1) ln1: x -> xn (fp32 + f16)
  ln_kernel<<<R, 256, 0, stream>>>(x, ln1_w, ln1_b, xn32, xn16, D);

  // 2) qkv = xn @ qkv_w + qkv_b  (f16 out)
  wcvt_kernel<<<(D * 3 * D) / 256, 256, 0, stream>>>(qkv_w, wT, D, 3 * D);
  gemm_wmma<0><<<dim3((3 * D) / 128, R / 128), 256, 0, stream>>>(
      xn16, wT, qkv_b, nullptr, nullptr, qkv16, R, 3 * D, D);

  // 3) flash attention -> o16
  attn_kernel<<<dim3(S / 16, B * H), 32, 0, stream>>>(qkv16, o16, S, D, H);

  // 4) h = xn + o @ attn_out_w + attn_out_b  (fp32)
  wcvt_kernel<<<(D * D) / 256, 256, 0, stream>>>(attn_out_w, wT, D, D);
  gemm_wmma<1><<<dim3(D / 128, R / 128), 256, 0, stream>>>(
      o16, wT, attn_out_b, xn32, h32, nullptr, R, D, D);

  // 5) ln2: h -> hn (fp32 + f16), reusing xn buffers
  ln_kernel<<<R, 256, 0, stream>>>(h32, ln2_w, ln2_b, xn32, xn16, D);

  // 6) a = gelu(hn @ c_fc_w + c_fc_b)  (f16 out, overlaps dead qkv16/o16)
  wcvt_kernel<<<(D * 4 * D) / 256, 256, 0, stream>>>(c_fc_w, wT, D, 4 * D);
  gemm_wmma<2><<<dim3((4 * D) / 128, R / 128), 256, 0, stream>>>(
      xn16, wT, c_fc_b, nullptr, nullptr, a16, R, 4 * D, D);

  // 7) out = hn + a @ c_proj_w + c_proj_b  (fp32 -> d_out)
  wcvt_kernel<<<(4 * D * D) / 256, 256, 0, stream>>>(c_proj_w, wT, 4 * D, D);
  gemm_wmma<1><<<dim3(D / 128, R / 128), 256, 0, stream>>>(
      a16, wT, c_proj_b, xn32, (float*)d_out, nullptr, R, D, 4 * D);
}
